// MultiHeadAttentionWithRPE_6442450944445
// MI455X (gfx1250) — matrix operand order repaired and checked by reading, not compiled, not run on hardware
//
#include <hip/hip_runtime.h>

// ---------------------------------------------------------------------------
// MultiHeadAttention with relative-position-bias — CDNA5 (gfx1250) WMMA path.
// B=32, N=512, D=512, H=8, DK=64, BINS=20, BIN_SIZE=5.0
// bf16 into the matrix units, f32 accumulation, f32 softmax.
// Round 4: TDM (tensor_load_to_lds, 6-arg toolchain form) stages the V^T tile,
// per-lane async-to-LDS stages the Q tile; attention phase 3 is fully LDS-fed.
// ---------------------------------------------------------------------------

#define B_    32
#define N_    512
#define D_    512
#define H_    8
#define DK_   64
#define BINS_ 20

#define USE_ASYNC_LDS 1

#ifndef __has_builtin
#define __has_builtin(x) 0
#endif
#if __has_builtin(__builtin_amdgcn_tensor_load_to_lds)
#define HAVE_TDM 1
#else
#define HAVE_TDM 0
#endif

typedef __attribute__((ext_vector_type(16))) __bf16        v16bf;
typedef __attribute__((ext_vector_type(8)))  __bf16        v8bf;
typedef __attribute__((ext_vector_type(8)))  float         v8f;
typedef __attribute__((ext_vector_type(4)))  unsigned int  v4u;
typedef __attribute__((ext_vector_type(8)))  int           v8i;
typedef __attribute__((ext_vector_type(4)))  int           v4i;

__device__ __forceinline__ unsigned short f2bf(float f) {
    union { float f; unsigned int u; } v; v.f = f;
    unsigned int u = v.u + 0x7FFFu + ((v.u >> 16) & 1u);   // round-to-nearest-even
    return (unsigned short)(u >> 16);
}

// A-fragment (16x32 bf16, M x K): lanes 0-15 hold row (lane&15), K = 0..15;
// lanes 16-31 hold K = 16..31 (two 16-byte chunks).
__device__ __forceinline__ v16bf load_a_frag(const unsigned short* rowptr, int kt, int hi) {
    const v8bf lo = *(const v8bf*)(rowptr + kt + hi * 16);
    const v8bf hh = *(const v8bf*)(rowptr + kt + 8 + hi * 16);
    return __builtin_shufflevector(lo, hh, 0,1,2,3,4,5,6,7,8,9,10,11,12,13,14,15);
}

// B-fragment (32x16 bf16, K x N): lane holds column (lane&15); lanes 0-15 carry
// K=0..15, lanes 16-31 carry K=16..31 — 16 consecutive K values -> 32B load.
__device__ __forceinline__ v16bf load_b_frag(const unsigned short* p) {
    return *(const v16bf*)p;
}

__device__ __forceinline__ v8f wmma_bf16(v16bf a, v16bf b, v8f c) {
    return __builtin_amdgcn_wmma_f32_16x16x32_bf16(false, a, false, b, (short)0, c, false, false);
}

// ---------------------------------------------------------------------------
// f32 -> bf16 conversion
// ---------------------------------------------------------------------------
__global__ __launch_bounds__(256)
void cvt_f32_bf16(const float* __restrict__ in, unsigned short* __restrict__ out, int n) {
    int i = blockIdx.x * 256 + threadIdx.x;
    if (i < n) out[i] = f2bf(in[i]);
}

// ---------------------------------------------------------------------------
// Y[M,512] = Xb[M,512] @ Wb[512,512]^T + bias   (M = 16384)
// MODE 0: bf16 row-major out   MODE 1: bf16 out transposed per batch [B,D,N]
// MODE 2: f32 row-major out
// Block = 256 threads (8 waves, 4x2); workgroup tile 256x128; wave tile 64x64.
// Per k-step: 16 b128 loads feed 16 WMMAs.
// ---------------------------------------------------------------------------
template <int MODE>
__global__ __launch_bounds__(256)
void gemm_bf16_512(const unsigned short* __restrict__ X,
                   const unsigned short* __restrict__ W,
                   const float* __restrict__ bias,
                   void* __restrict__ Yv)
{
    const int K    = D_;
    const int lane = threadIdx.x & 31;
    const int w    = threadIdx.x >> 5;
    const int col  = lane & 15;
    const int hi   = lane >> 4;

    const int m0 = blockIdx.x * 256 + (w & 3) * 64;
    const int n0 = blockIdx.y * 128 + (w >> 2) * 64;

    v8f acc[4][4] = {};

    const unsigned short* xr[4];
    const unsigned short* wr[4];
    #pragma unroll
    for (int r = 0; r < 4; ++r) xr[r] = X + (size_t)(m0 + r * 16 + col) * K;
    #pragma unroll
    for (int c = 0; c < 4; ++c) wr[c] = W + (size_t)(n0 + c * 16 + col) * K;

    for (int kt = 0; kt < K; kt += 32) {
        v16bf a[4], bf[4];
        #pragma unroll
        for (int r = 0; r < 4; ++r) a[r] = load_a_frag(xr[r], kt, hi);
        #pragma unroll
        for (int c = 0; c < 4; ++c) bf[c] = load_b_frag(wr[c] + kt + hi * 16);
        #pragma unroll
        for (int r = 0; r < 4; ++r)
            #pragma unroll
            for (int c = 0; c < 4; ++c)
                acc[r][c] = wmma_bf16(a[r], bf[c], acc[r][c]);
        if (kt + 32 < K) {
            __builtin_prefetch(xr[0] + kt + 32, 0, 0);   // global_prefetch_b8
            __builtin_prefetch(wr[0] + kt + 32, 0, 0);
        }
    }

    #pragma unroll
    for (int r = 0; r < 4; ++r) {
        #pragma unroll
        for (int cb = 0; cb < 4; ++cb) {
            const int gn = n0 + cb * 16 + col;
            const float bv = bias[gn];
            #pragma unroll
            for (int i = 0; i < 8; ++i) {
                const int gm = m0 + r * 16 + i + hi * 8;     // row of C tile
                const float val = acc[r][cb][i] + bv;
                if (MODE == 0) {
                    ((unsigned short*)Yv)[(size_t)gm * D_ + gn] = f2bf(val);
                } else if (MODE == 1) {
                    const int bb = gm >> 9;                  // gm / N_
                    const int nn = gm & (N_ - 1);
                    ((unsigned short*)Yv)[(size_t)bb * D_ * N_ + (size_t)gn * N_ + nn] = f2bf(val);
                } else {
                    ((float*)Yv)[(size_t)gm * D_ + gn] = val;
                }
            }
        }
    }
}

// ---------------------------------------------------------------------------
// Attention: one block per (b, h, 64-query tile). 256 threads = 8 waves.
// Phase 0a: async-stage 64x64 bf16 Q tile into LDS (ASYNCcnt path).
// Phase 0b: TDM-stage 64x512 bf16 V^T tile into LDS (TENSORcnt path, wave 0).
// Phase 1 : S = Q K^T * 1/8 + dist_weights[bin(dist)]   -> LDS (64x512 f32)
// Phase 2 : row softmax (4 threads per row), pack P bf16 -> LDS (64x512 bf16)
// Phase 3 : O = P V, all operands from LDS.
// Dynamic LDS layout (bytes):
//   S   @ 0        64*512*4 = 131072
//   P   @ 131072   64*512*2 =  65536
//   dwl @ 196608              128
//   red @ 196736              1024
//   qsh @ 197760   64*64*2  =   8192
//   vsh @ 205952   64*512*2 =  65536    total 271488 (<= 320KB WGP LDS)
// ---------------------------------------------------------------------------
#define SMEM_S    0
#define SMEM_P    131072
#define SMEM_DWL  196608
#define SMEM_RED  196736
#define SMEM_QSH  197760
#define SMEM_VSH  205952
#define SMEM_TOT  271488

__global__ __launch_bounds__(256)
void attn_kernel(const unsigned short* __restrict__ Qb,
                 const unsigned short* __restrict__ Kb,
                 const unsigned short* __restrict__ Vt,
                 const float* __restrict__ dist,
                 const float* __restrict__ dw,
                 unsigned short* __restrict__ Ob)
{
    extern __shared__ char smem[];
    float*          S   = (float*)(smem + SMEM_S);            // 64x512 f32
    unsigned short* P   = (unsigned short*)(smem + SMEM_P);   // 64x512 bf16
    float*          dwl = (float*)(smem + SMEM_DWL);          // 32 f32
    float*          red = (float*)(smem + SMEM_RED);          // 256 f32
    unsigned short* qsh = (unsigned short*)(smem + SMEM_QSH); // 64x64 bf16
    unsigned short* vsh = (unsigned short*)(smem + SMEM_VSH); // 64x512 bf16 (dk-major)

    const int tid  = threadIdx.x;
    const int lane = tid & 31;
    const int w    = tid >> 5;
    const int col  = lane & 15;
    const int hi   = lane >> 4;

    const int bh   = blockIdx.x >> 3;       // / (N/64)
    const int qblk = blockIdx.x & 7;
    const int b    = bh >> 3;               // / H
    const int h    = bh & (H_ - 1);
    const int q0   = qblk * 64;

    if (tid <= BINS_) dwl[tid] = dw[tid];

    // ---- Phase 0a: per-lane async DMA of Q tile (64 rows x 64 dk, bf16) ----
    {
        const int r     = tid >> 2;                 // 0..63   query row
        const int chunk = tid & 3;                  // 0..3    32B chunk
        const unsigned short* gsrc =
            Qb + ((size_t)b * N_ + q0 + r) * D_ + h * DK_ + chunk * 16;
        const unsigned int lds_off = SMEM_QSH + r * 128 + chunk * 32;
#if USE_ASYNC_LDS
        asm volatile(
            "global_load_async_to_lds_b128 %0, %1, off\n\t"
            "global_load_async_to_lds_b128 %0, %1, off offset:16"
            :: "v"(lds_off), "v"(gsrc) : "memory");
#else
        *(v8bf*)(smem + lds_off)      = *(const v8bf*)(gsrc);
        *(v8bf*)(smem + lds_off + 16) = *(const v8bf*)(gsrc + 8);
#endif
    }

    // ---- Phase 0b: TDM DMA of V^T tile (64 dk rows x 512 keys, bf16) ----
    const unsigned short* vtile = Vt + (size_t)b * D_ * N_ + (size_t)(h * DK_) * N_;
#if HAVE_TDM
    if (w == 0) {
        const unsigned long long ga = (unsigned long long)(uintptr_t)vtile;
        v4u g0;
        g0.x = 1u;                                              // count=1, no gather
        g0.y = (unsigned)SMEM_VSH;                              // lds_addr (bytes)
        g0.z = (unsigned)(ga & 0xFFFFFFFFu);                    // global_addr[31:0]
        g0.w = (unsigned)((ga >> 32) & 0x01FFFFFFu) | (2u << 30);  // addr[56:32] | type=2
        v8i g1;
        g1[0] = 0x00010000;                                     // data_size=1 -> 2B elems
        g1[1] = (int)(512u << 16);                              // tensor_dim0 = 512
        g1[2] = (int)(64u  << 16);                              // tensor_dim1 = 64
        g1[3] = (int)(512u << 16);                              // tile_dim0  = 512
        g1[4] = 64;                                             // tile_dim1  = 64
        g1[5] = 512;                                            // tensor_dim0_stride = 512
        g1[6] = 0;
        g1[7] = 0;
        const v4i gz4 = {};
        const v8i gz8 = {};
        // 6-arg toolchain form: (g0, g1, g2, g3, g4, cpol)
        __builtin_amdgcn_tensor_load_to_lds(g0, g1, gz4, gz4, gz8, 0);
#if __has_builtin(__builtin_amdgcn_s_wait_tensorcnt)
        __builtin_amdgcn_s_wait_tensorcnt(0);
#else
        asm volatile("s_wait_tensorcnt 0x0" ::: "memory");
#endif
    }
#else
    {   // Fallback: per-lane async staging of V^T (256B per thread).
        const int d       = tid >> 2;               // 0..63 dk row
        const int quarter = tid & 3;                // 0..3  256B chunk of the row
        const unsigned short* gsrc = vtile + (size_t)d * N_ + quarter * 128;
        unsigned int lds_off = SMEM_VSH + d * 1024 + quarter * 256;
        #pragma unroll
        for (int i = 0; i < 8; ++i) {
            asm volatile(
                "global_load_async_to_lds_b128 %0, %1, off\n\t"
                "global_load_async_to_lds_b128 %0, %1, off offset:16"
                :: "v"(lds_off), "v"(gsrc) : "memory");
            lds_off += 32; gsrc += 16;
        }
    }
#endif
#if USE_ASYNC_LDS
    asm volatile("s_wait_asynccnt 0x0" ::: "memory");
#endif
    __syncthreads();

    // ---- Phase 1: wave w computes S strip for keys [w*64, w*64+64) ----
    {
        const unsigned short* kbase = Kb + ((size_t)b * N_ + w * 64) * D_ + h * DK_;
        const float scale = 0.125f;         // 1/sqrt(DK)

        #pragma unroll
        for (int qr = 0; qr < 4; ++qr) {
            const unsigned short* qrow = qsh + (size_t)(qr * 16 + col) * DK_;
            v16bf a0 = load_a_frag(qrow, 0,  hi);   // ds_load path
            v16bf a1 = load_a_frag(qrow, 32, hi);
            #pragma unroll
            for (int kc = 0; kc < 4; ++kc) {
                const unsigned short* krow = kbase + (size_t)(kc * 16 + col) * D_;
                v16bf b0 = load_b_frag(krow + 0  + hi * 16);
                v16bf b1 = load_b_frag(krow + 32 + hi * 16);
                v8f acc = {};
                acc = wmma_bf16(a0, b0, acc);
                acc = wmma_bf16(a1, b1, acc);
                #pragma unroll
                for (int i = 0; i < 8; ++i) {
                    const int row = qr * 16 + i + hi * 8;
                    const int key = w * 64 + kc * 16 + col;
                    const float dv = dist[(size_t)b * N_ * N_ + (size_t)(q0 + row) * N_ + key];
                    int bin = (int)(dv * 0.2f);            // / BIN_SIZE (=5)
                    bin = bin < 0 ? 0 : (bin > BINS_ ? BINS_ : bin);
                    S[row * 512 + key] = acc[i] * scale + dwl[bin];
                }
            }
        }
    }
    __syncthreads();

    // ---- Phase 2: softmax, 4 threads per row ----
    {
        const int r    = tid >> 2;
        const int part = tid & 3;
        const int k0   = part * 128;
        float m = -3.0e38f;
        for (int k = 0; k < 128; ++k) m = fmaxf(m, S[r * 512 + k0 + k]);
        red[tid] = m;
        __syncthreads();
        const float rm = fmaxf(fmaxf(red[r * 4 + 0], red[r * 4 + 1]),
                               fmaxf(red[r * 4 + 2], red[r * 4 + 3]));
        float s = 0.0f;
        for (int k = 0; k < 128; ++k) {
            float e = __expf(S[r * 512 + k0 + k] - rm);
            S[r * 512 + k0 + k] = e;
            s += e;
        }
        __syncthreads();
        red[tid] = s;
        __syncthreads();
        const float inv = 1.0f / (red[r * 4 + 0] + red[r * 4 + 1] +
                                  red[r * 4 + 2] + red[r * 4 + 3]);
        for (int k = 0; k < 128; ++k)
            P[r * 512 + k0 + k] = f2bf(S[r * 512 + k0 + k] * inv);
    }
    __syncthreads();

    // ---- Phase 3: O = P @ V, entirely LDS-fed ----
    {
        const int qr0 = (w & 3) * 16;
        const int d0  = (w >> 2) * 32;
        v8f o0 = {}, o1 = {};
        const unsigned short* prow  = P + (size_t)(qr0 + col) * 512;
        const unsigned short* vrow0 = vsh + (size_t)(d0 + col) * 512;        // dk-major LDS tile
        const unsigned short* vrow1 = vsh + (size_t)(d0 + 16 + col) * 512;

        for (int kt = 0; kt < N_; kt += 32) {
            v16bf a  = load_a_frag(prow, kt, hi);        // ds_load
            v16bf b0 = load_b_frag(vrow0 + kt + hi * 16); // ds_load
            v16bf b1 = load_b_frag(vrow1 + kt + hi * 16);
            o0 = wmma_bf16(a, b0, o0);
            o1 = wmma_bf16(a, b1, o1);
        }
        v8f oo[2] = { o0, o1 };
        #pragma unroll
        for (int cb = 0; cb < 2; ++cb) {
            #pragma unroll
            for (int i = 0; i < 8; ++i) {
                const int row = qr0 + i + hi * 8;
                const int gd  = h * DK_ + d0 + cb * 16 + col;
                Ob[((size_t)b * N_ + q0 + row) * D_ + gd] = f2bf(oo[cb][i]);
            }
        }
    }
}

// ---------------------------------------------------------------------------
// Host-side launch
// ---------------------------------------------------------------------------
extern "C" void kernel_launch(void* const* d_in, const int* in_sizes, int n_in,
                              void* d_out, int out_size, void* d_ws, size_t ws_size,
                              hipStream_t stream) {
    (void)in_sizes; (void)n_in; (void)out_size; (void)ws_size;

    const float* x    = (const float*)d_in[0];
    const float* dist = (const float*)d_in[1];
    const float* Wq   = (const float*)d_in[2];
    const float* bq   = (const float*)d_in[3];
    const float* Wk   = (const float*)d_in[4];
    const float* bk   = (const float*)d_in[5];
    const float* Wv   = (const float*)d_in[6];
    const float* bv   = (const float*)d_in[7];
    const float* Wo   = (const float*)d_in[8];
    const float* bo   = (const float*)d_in[9];
    const float* dw   = (const float*)d_in[10];
    float* out = (float*)d_out;

    const size_t XN = (size_t)B_ * N_ * D_;     // 8,388,608 elements
    const size_t WN = (size_t)D_ * D_;          // 262,144 elements

    char* ws = (char*)d_ws;
    unsigned short* xb  = (unsigned short*)(ws);
    unsigned short* Wqb = (unsigned short*)(ws + 16777216);
    unsigned short* Wkb = (unsigned short*)(ws + 16777216 + 1 * 524288);
    unsigned short* Wvb = (unsigned short*)(ws + 16777216 + 2 * 524288);
    unsigned short* Wob = (unsigned short*)(ws + 16777216 + 3 * 524288);
    unsigned short* Qb  = (unsigned short*)(ws + 18874368);
    unsigned short* Kb  = (unsigned short*)(ws + 18874368 + 1 * 16777216);
    unsigned short* Vt  = (unsigned short*)(ws + 18874368 + 2 * 16777216);  // [B, D, N]
    unsigned short* Ob  = (unsigned short*)(ws + 18874368 + 3 * 16777216);

    // 1) Convert x and weights to bf16
    cvt_f32_bf16<<<(int)((XN + 255) / 256), 256, 0, stream>>>(x, xb, (int)XN);
    cvt_f32_bf16<<<(int)((WN + 255) / 256), 256, 0, stream>>>(Wq, Wqb, (int)WN);
    cvt_f32_bf16<<<(int)((WN + 255) / 256), 256, 0, stream>>>(Wk, Wkb, (int)WN);
    cvt_f32_bf16<<<(int)((WN + 255) / 256), 256, 0, stream>>>(Wv, Wvb, (int)WN);
    cvt_f32_bf16<<<(int)((WN + 255) / 256), 256, 0, stream>>>(Wo, Wob, (int)WN);

    // 2) Projections: M = B*N = 16384 rows; grid = (16384/256, 512/128)
    dim3 ggrid(64, 4);
    gemm_bf16_512<0><<<ggrid, 256, 0, stream>>>(xb, Wqb, bq, Qb);
    gemm_bf16_512<0><<<ggrid, 256, 0, stream>>>(xb, Wkb, bk, Kb);
    gemm_bf16_512<1><<<ggrid, 256, 0, stream>>>(xb, Wvb, bv, Vt);   // store V^T

    // 3) Attention: B*H*(N/64) = 2048 blocks, ~265 KB dynamic LDS each
    attn_kernel<<<B_ * H_ * (N_ / 64), 256, SMEM_TOT, stream>>>(Qb, Kb, Vt, dist, dw, Ob);

    // 4) Output projection (f32 result straight to d_out)
    gemm_bf16_512<2><<<ggrid, 256, 0, stream>>>(Ob, Wob, bo, out);
}
